// AttentionLayer_68229850464552
// MI455X (gfx1250) — matrix operand order repaired and checked by reading, not compiled, not run on hardware
//
#include <hip/hip_runtime.h>
#include <hip/hip_bf16.h>

// ---------------------------------------------------------------------------
// CDNA5 (gfx1250) attention layer: bf16 WMMA GEMMs + flash-style attention.
// wave32; v_wmma_f32_16x16x32_bf16 for all matmuls; double-buffered
// async-to-LDS staging (GLOBAL_LOAD_ASYNC_TO_LDS_B128 / ASYNCcnt) pipelined
// against the WMMA compute.
// ---------------------------------------------------------------------------

typedef __attribute__((ext_vector_type(16))) __bf16 v16bf;
typedef __attribute__((ext_vector_type(8)))  float  v8f;

union Frag {
    v16bf v;
    unsigned int u[8];
};

#define WMMA_BF16(A, B, C) \
    __builtin_amdgcn_wmma_f32_16x16x32_bf16(false, (A), false, (B), (short)0, (C), false, false)

#if __has_builtin(__builtin_amdgcn_global_load_async_to_lds_b128)
#define USE_ASYNC_LDS 1
#endif

// Pointer types matching the builtin's signature: int4 vectors in the
// global (AS1) and LDS (AS3) address spaces.
typedef int v4i_vec __attribute__((vector_size(16)));
typedef __attribute__((address_space(1))) v4i_vec* g_b128_ptr;
typedef __attribute__((address_space(3))) v4i_vec* l_b128_ptr;

// Copy 16 bytes global -> LDS. Async path uses the CDNA5 direct-to-LDS DMA
// (no VGPR round trip, tracked by ASYNCcnt); fallback is load+ds_store.
__device__ __forceinline__ void cp_b128(unsigned int* lds, const uint4* g)
{
#ifdef USE_ASYNC_LDS
    __builtin_amdgcn_global_load_async_to_lds_b128(
        (g_b128_ptr)(uintptr_t)g,
        (l_b128_ptr)(unsigned)(uintptr_t)lds,
        0, 0);
#else
    uint4 t = *g;
    lds[0] = t.x; lds[1] = t.y; lds[2] = t.z; lds[3] = t.w;
#endif
}

__device__ __forceinline__ void wait_async()
{
#ifdef USE_ASYNC_LDS
    asm volatile("s_wait_asynccnt 0x0" ::: "memory");
#endif
}

// ---------------------------------------------------------------------------
// fp32 -> bf16 convert (grid-stride)
// ---------------------------------------------------------------------------
__global__ __launch_bounds__(256) void cvt_bf16_kernel(
    const float* __restrict__ in, __bf16* __restrict__ out, int n)
{
    int i = blockIdx.x * blockDim.x + threadIdx.x;
    int stride = gridDim.x * blockDim.x;
    for (; i < n; i += stride) out[i] = (__bf16)in[i];
}

// ---------------------------------------------------------------------------
// C[M,N] = A[M,K] (bf16, row-major) x W[N,K]^T (bf16, row-major) + bias
// Tile: 128x128, K-step 32, double-buffered: async stage of tile k+1 is
// issued before the WMMA work on tile k (one barrier per iteration).
// 256 threads = 8 waves in a 2(M) x 4(N) grid; each wave: 64x32 sub-tile
// as 4x2 WMMA accumulators. LDS row stride 20 dwords (80 B: 16B-aligned
// for async b128; 20*lm mod 64 is a permutation -> conflict-free gathers).
// ---------------------------------------------------------------------------
#define GSTRIDE 20

template <bool OUT_BF16>
__global__ __launch_bounds__(256) void gemm_wt_kernel(
    const __bf16* __restrict__ A, const __bf16* __restrict__ W,
    const float* __restrict__ bias, void* __restrict__ Cout,
    int M, int N, int K)
{
    __shared__ alignas(16) unsigned int Al[2][128 * GSTRIDE];
    __shared__ alignas(16) unsigned int Bl[2][128 * GSTRIDE];

    const int tid  = threadIdx.x;
    const int lane = tid & 31;
    const int wave = __builtin_amdgcn_readfirstlane(tid >> 5);  // scalar
    const int wm   = wave & 1;     // 0..1  (64-row group)
    const int wn   = wave >> 1;    // 0..3  (32-col group)
    const int lm   = lane & 15;
    const int lh   = lane >> 4;

    const int m0 = blockIdx.y * 128;
    const int n0 = blockIdx.x * 128;

    v8f acc[4][2];
#pragma unroll
    for (int i = 0; i < 4; ++i)
#pragma unroll
        for (int j = 0; j < 2; ++j) acc[i][j] = (v8f){};

    const int lrow = tid >> 1;          // 0..127
    const int lseg = (tid & 1) * 8;     // dword offset within the 16-dw row

    const __bf16* gA = A + (size_t)(m0 + lrow) * K + lseg * 2;
    const __bf16* gW = W + (size_t)(n0 + lrow) * K + lseg * 2;

    auto stage = [&](int buf, int k0) {
        cp_b128(&Al[buf][lrow * GSTRIDE + lseg],     (const uint4*)(gA + k0));
        cp_b128(&Al[buf][lrow * GSTRIDE + lseg + 4], (const uint4*)(gA + k0) + 1);
        cp_b128(&Bl[buf][lrow * GSTRIDE + lseg],     (const uint4*)(gW + k0));
        cp_b128(&Bl[buf][lrow * GSTRIDE + lseg + 4], (const uint4*)(gW + k0) + 1);
        if (k0 + 32 < K) {   // prefetch one more tile ahead of the pipeline
            __builtin_prefetch(gA + k0 + 32, 0, 1);
            __builtin_prefetch(gW + k0 + 32, 0, 1);
        }
    };

    const int nIter = K >> 5;
    stage(0, 0);
    int buf = 0;

    for (int it = 0; it < nIter; ++it) {
        wait_async();
        __syncthreads();
        if (it + 1 < nIter) stage(buf ^ 1, (it + 1) * 32);

        Frag afr[4];
#pragma unroll
        for (int mi = 0; mi < 4; ++mi) {
            const unsigned int* row = &Al[buf][(wm * 64 + mi * 16 + lm) * GSTRIDE];
#pragma unroll
            for (int v = 0; v < 8; ++v) {
                int dw = (v < 4) ? (lh * 4 + v) : (8 + lh * 4 + (v - 4));
                afr[mi].u[v] = row[dw];
            }
        }
        Frag bfr[2];
#pragma unroll
        for (int nj = 0; nj < 2; ++nj) {
            const unsigned int* row = &Bl[buf][(wn * 32 + nj * 16 + lm) * GSTRIDE];
#pragma unroll
            for (int v = 0; v < 8; ++v) bfr[nj].u[v] = row[(v < 4) ? (lh * 4 + v) : (8 + lh * 4 + (v - 4))];
        }
#pragma unroll
        for (int mi = 0; mi < 4; ++mi)
#pragma unroll
            for (int nj = 0; nj < 2; ++nj)
                acc[mi][nj] = WMMA_BF16(afr[mi].v, bfr[nj].v, acc[mi][nj]);

        buf ^= 1;
    }

    // epilogue: bias + store (C layout: VGPR r -> row r + 8*lh, col = lm)
#pragma unroll
    for (int mi = 0; mi < 4; ++mi)
#pragma unroll
        for (int nj = 0; nj < 2; ++nj) {
            int ng = n0 + wn * 32 + nj * 16 + lm;
            float bv = bias ? bias[ng] : 0.0f;
#pragma unroll
            for (int r = 0; r < 8; ++r) {
                int mg = m0 + wm * 64 + mi * 16 + r + 8 * lh;
                float val = acc[mi][nj][r] + bv;
                if (OUT_BF16)
                    ((__bf16*)Cout)[(size_t)mg * N + ng] = (__bf16)val;
                else
                    ((float*)Cout)[(size_t)mg * N + ng] = val;
            }
        }
}

// ---------------------------------------------------------------------------
// RoPE (in-place on bf16 qkv). qkv row layout: 16 heads x (q128|k128|v128).
// Rotates first 32 dims of q and k (half = 16).
// One thread per (row, head, d<16).
// ---------------------------------------------------------------------------
__global__ __launch_bounds__(256) void rope_kernel(
    __bf16* __restrict__ qkv, int BS, int S)
{
    int idx = blockIdx.x * blockDim.x + threadIdx.x;
    int d   = idx & 15;
    int h   = (idx >> 4) & 15;
    int row = idx >> 8;
    if (row >= BS) return;
    int s = row % S;

    float theta = __powf(10000.0f, -(float)d / 16.0f);
    float ang = (float)s * theta;
    float c = __cosf(ang), sn = __sinf(ang);

    __bf16* q = qkv + (size_t)row * 6144 + h * 384 + d;
    float x0 = (float)q[0], x1 = (float)q[16];
    q[0]  = (__bf16)(x0 * c - x1 * sn);
    q[16] = (__bf16)(x1 * c + x0 * sn);

    __bf16* k = q + 128;
    x0 = (float)k[0]; x1 = (float)k[16];
    k[0]  = (__bf16)(x0 * c - x1 * sn);
    k[16] = (__bf16)(x1 * c + x0 * sn);
}

// ---------------------------------------------------------------------------
// Flash-style causal attention. grid = (S/128, H, B), 256 threads.
// Each wave owns 16 query rows. Double-buffered KV staging: K tile goes
// through async-to-LDS; V tile is loaded to VGPRs one iteration ahead and
// scattered (transposed) to LDS at the top of the next iteration, so both
// copies overlap the WMMA compute.
// ---------------------------------------------------------------------------
#define KSTRIDE 68   // dwords per K row in LDS (272 B: 16B-aligned)

union VCarry { uint4 q[2]; __bf16 h[16]; };

__global__ __launch_bounds__(256) void flash_attn_kernel(
    const __bf16* __restrict__ qkv, __bf16* __restrict__ ctx,
    int Bn, int S)
{
    __shared__ alignas(16) unsigned int Kl[2][32 * KSTRIDE];
    __shared__ alignas(16) __bf16 Vt[2][128][34];   // transposed V: [dim][key]
    __shared__ alignas(16) __bf16 Pl[8][16][32];    // per-wave P (C->A relayout)

    const int tid  = threadIdx.x;
    const int lane = tid & 31;
    const int wave = __builtin_amdgcn_readfirstlane(tid >> 5);  // scalar
    const int lm   = lane & 15;
    const int lh   = lane >> 4;

    const int b  = blockIdx.z;
    const int h  = blockIdx.y;
    const int m0 = blockIdx.x * 128;
    const int mW = m0 + wave * 16;            // this wave's first query row

    const float scale = 0.08838834764831845f; // 1/sqrt(128)
    const size_t rs = 6144;
    const int colQ = h * 384, colK = colQ + 128, colV = colQ + 256;

    // --- Q fragments (4 chunks of K-dim 32), A-matrix layout, loaded once ---
    Frag qf[4];
    {
        const unsigned int* qd =
            (const unsigned int*)(qkv + (size_t)(b * S + mW + lm) * rs + colQ);
#pragma unroll
        for (int f = 0; f < 4; ++f)
#pragma unroll
            for (int v = 0; v < 8; ++v) {
                int dw = (v < 4) ? (lh * 4 + v) : (8 + lh * 4 + (v - 4));
                qf[f].u[v] = qd[f * 16 + dw];
            }
    }

    v8f o[8];
#pragma unroll
    for (int t = 0; t < 8; ++t) o[t] = (v8f){};
    float mi_[8], li_[8];
#pragma unroll
    for (int r = 0; r < 8; ++r) { mi_[r] = -1e30f; li_[r] = 0.0f; }

    // KV staging indices
    const int svrow = tid >> 3;           // 0..31 (key within tile)
    const int svseg = (tid & 7) * 8;      // dword segment within 64-dw row

    // Stage tile at j0 into buffer `buf`: K via async DMA, V into VGPRs.
    auto stage = [&](int buf, int j0) -> VCarry {
        const uint4* gk =
            (const uint4*)(qkv + (size_t)(b * S + j0 + svrow) * rs + colK + svseg * 2);
        cp_b128(&Kl[buf][svrow * KSTRIDE + svseg],     gk);
        cp_b128(&Kl[buf][svrow * KSTRIDE + svseg + 4], gk + 1);
        const uint4* gv =
            (const uint4*)(qkv + (size_t)(b * S + j0 + svrow) * rs + colV + svseg * 2);
        VCarry vc;
        vc.q[0] = gv[0];
        vc.q[1] = gv[1];
        return vc;
    };

    const int nIter = (m0 + 128) >> 5;
    VCarry vc = stage(0, 0);
    int buf = 0;

    for (int it = 0; it < nIter; ++it) {
        const int j0 = it * 32;

        // scatter this tile's V (transposed) into LDS, then publish
#pragma unroll
        for (int e = 0; e < 16; ++e) Vt[buf][svseg * 2 + e][svrow] = vc.h[e];
        wait_async();
        __syncthreads();

        if (it + 1 < nIter) vc = stage(buf ^ 1, j0 + 32);

        if (j0 <= mW + 15) {   // wave-uniform: any unmasked key in this tile
            // ---- S = Q K^T for two 16-key sub-tiles ----
            v8f st[2];
            st[0] = (v8f){}; st[1] = (v8f){};
#pragma unroll
            for (int jt = 0; jt < 2; ++jt) {
                const unsigned int* krow = &Kl[buf][(jt * 16 + lm) * KSTRIDE];
#pragma unroll
                for (int f = 0; f < 4; ++f) {
                    Frag bf_;
#pragma unroll
                    for (int v = 0; v < 8; ++v) bf_.u[v] = krow[f * 16 + ((v < 4) ? (lh * 4 + v) : (8 + lh * 4 + (v - 4)))];
                    st[jt] = WMMA_BF16(qf[f].v, bf_.v, st[jt]);
                }
            }

            // ---- mask + scale + online softmax stats ----
            float pv0[8], pv1[8], alpha[8];
#pragma unroll
            for (int r = 0; r < 8; ++r) {
                int mg  = mW + r + 8 * lh;
                int jg0 = j0 + lm;
                int jg1 = j0 + 16 + lm;
                float s0 = (jg0 <= mg) ? st[0][r] * scale : -10000.0f;
                float s1 = (jg1 <= mg) ? st[1][r] * scale : -10000.0f;
                float mx = fmaxf(s0, s1);
#pragma unroll
                for (int off = 1; off < 16; off <<= 1)
                    mx = fmaxf(mx, __shfl_xor(mx, off, 16));
                float mn = fmaxf(mi_[r], mx);
                float a  = __expf(mi_[r] - mn);
                float p0 = __expf(s0 - mn);
                float p1 = __expf(s1 - mn);
                float rsum = p0 + p1;
#pragma unroll
                for (int off = 1; off < 16; off <<= 1)
                    rsum += __shfl_xor(rsum, off, 16);
                li_[r] = li_[r] * a + rsum;
                mi_[r] = mn;
                alpha[r] = a; pv0[r] = p0; pv1[r] = p1;
            }
#pragma unroll
            for (int t = 0; t < 8; ++t)
#pragma unroll
                for (int r = 0; r < 8; ++r) o[t][r] *= alpha[r];

            // ---- C-layout P -> LDS -> A-layout fragment ----
#pragma unroll
            for (int r = 0; r < 8; ++r) {
                int rr = r + 8 * lh;
                Pl[wave][rr][lm]      = (__bf16)pv0[r];
                Pl[wave][rr][16 + lm] = (__bf16)pv1[r];
            }
            Frag pf;
            {
                const unsigned int* prow = (const unsigned int*)&Pl[wave][lm][0];
#pragma unroll
                for (int v = 0; v < 8; ++v) {
                    int dw = (v < 4) ? (lh * 4 + v) : (8 + lh * 4 + (v - 4));
                    pf.u[v] = prow[dw];
                }
            }

            // ---- O += P x V over 8 output-dim sub-tiles ----
#pragma unroll
            for (int t = 0; t < 8; ++t) {
                Frag vf;
                const unsigned int* vrow = (const unsigned int*)&Vt[buf][t * 16 + lm][0];
#pragma unroll
                for (int v = 0; v < 8; ++v) vf.u[v] = vrow[(v < 4) ? (lh * 4 + v) : (8 + lh * 4 + (v - 4))];
                o[t] = WMMA_BF16(pf.v, vf.v, o[t]);
            }
        }
        buf ^= 1;
    }

    // ---- normalize + store context (bf16, [B*S, 2048], head-major cols) ----
#pragma unroll
    for (int t = 0; t < 8; ++t)
#pragma unroll
        for (int r = 0; r < 8; ++r) {
            int mg = mW + r + 8 * lh;
            float val = o[t][r] / li_[r];
            ctx[(size_t)(b * S + mg) * 2048 + h * 128 + t * 16 + lm] = (__bf16)val;
        }
}

// ---------------------------------------------------------------------------
// Launcher
// ---------------------------------------------------------------------------
extern "C" void kernel_launch(void* const* d_in, const int* in_sizes, int n_in,
                              void* d_out, int out_size, void* d_ws, size_t ws_size,
                              hipStream_t stream)
{
    (void)in_sizes; (void)n_in; (void)out_size; (void)ws_size;

    const float* x     = (const float*)d_in[0];   // [2,2048,2048]
    const float* W_qkv = (const float*)d_in[1];   // [6144,2048]
    const float* b_qkv = (const float*)d_in[2];   // [6144]
    const float* W_out = (const float*)d_in[3];   // [2048,2048]
    const float* b_out = (const float*)d_in[4];   // [2048]
    float* out = (float*)d_out;                   // [2,2048,2048]

    const int Bn = 2, S = 2048, D = 2048;
    const int M = Bn * S;            // 4096
    const int N3 = 3 * D;            // 6144

    char* ws = (char*)d_ws;
    __bf16* xb    = (__bf16*)ws; ws += (size_t)M * D * 2;        // 16.8 MB
    __bf16* wqkvb = (__bf16*)ws; ws += (size_t)N3 * D * 2;       // 25.2 MB
    __bf16* woutb = (__bf16*)ws; ws += (size_t)D * D * 2;        //  8.4 MB
    __bf16* qkvb  = (__bf16*)ws; ws += (size_t)M * N3 * 2;       // 50.3 MB
    __bf16* ctx   = (__bf16*)ws;                                 // 16.8 MB

    // 1) fp32 -> bf16 converts
    cvt_bf16_kernel<<<4096, 256, 0, stream>>>(x, xb, M * D);
    cvt_bf16_kernel<<<4096, 256, 0, stream>>>(W_qkv, wqkvb, N3 * D);
    cvt_bf16_kernel<<<4096, 256, 0, stream>>>(W_out, woutb, D * D);

    // 2) qkv = x @ W_qkv^T + b_qkv  (bf16 out)
    gemm_wt_kernel<true><<<dim3(N3 / 128, M / 128), 256, 0, stream>>>(
        xb, wqkvb, b_qkv, qkvb, M, N3, D);

    // 3) RoPE on q,k (in place)
    rope_kernel<<<M, 256, 0, stream>>>(qkvb, M, S);

    // 4) flash attention -> ctx
    flash_attn_kernel<<<dim3(S / 128, 16, Bn), 256, 0, stream>>>(qkvb, ctx, Bn, S);

    // 5) out = ctx @ W_out^T + b_out  (fp32 out)
    gemm_wt_kernel<false><<<dim3(D / 128, M / 128), 256, 0, stream>>>(
        ctx, woutb, b_out, out, M, D, D);
}
